// PhiLang_89936615178978
// MI455X (gfx1250) — matrix operand order repaired and checked by reading, not compiled, not run on hardware
//
#include <hip/hip_runtime.h>

// ---------------------------------------------------------------------------
// CDNA5 (gfx1250) implementation of the PhiLang heterogeneous graph transformer.
// All GEMMs run on v_wmma_f32_16x16x32_bf16 (f32 accumulate). Edge attention is
// algebraically refactored to avoid materializing the 268MB e = ea@We tensor:
//   logits:  q.e  = ea . (We^T q)      (qWe precomputed, batched WMMA GEMM)
//   message: sum(a*e) = (sum(a*ea)) @ We
// 'guide'/'attend' edge types have one edge per destination -> softmax == 1,
// so their message is a single broadcast row; Q/K projections are dead code.
//
// Bounds policy: out-of-range rows/cols are CLAMPED at load (unconditional,
// vectorized b64/b128 loads) and masked only at the store epilogue. N and K
// are multiples of 64/32 for every GEMM in this model.
// ---------------------------------------------------------------------------

typedef __attribute__((ext_vector_type(16))) __bf16 v16bf;
typedef __attribute__((ext_vector_type(8)))  float  v8f;

#define F_BIAS   1
#define F_GELU   2
#define F_ACCUM  4
#define F_TRANSB 8

__device__ __forceinline__ float gelu_tanh(float x) {
    float x3 = x * x * x;
    return 0.5f * x * (1.0f + tanhf(0.7978845608028654f * (x + 0.044715f * x3)));
}

// A fragment: 16x32 bf16, M on lanes (lane%16), K packed in VGPRs per ISA 7.12.2.
// arow = pre-clamped row base pointer for this lane; hb = 0 or 8 (lane half).
__device__ __forceinline__ v16bf frag_a_row(const float* __restrict__ arow,
                                            int k0, int hb) {
    v16bf f;
#pragma unroll
    for (int v = 0; v < 8; ++v) {
        int k = k0 + ((v >> 2) << 4) + hb + ((v & 3) << 1);
        float2 x = *(const float2*)(arow + k);   // 8B aligned: lda even, k even
        f[2 * v]     = (__bf16)x.x;
        f[2 * v + 1] = (__bf16)x.y;
    }
    return f;
}

// B fragment: 32x16 bf16, N on lanes (lane%16), K packed in VGPRs (table order).
// br = &B[(k0+hb)*ldb + col0 + lane%16]; strided scalar loads.
__device__ __forceinline__ v16bf frag_b_row(const float* __restrict__ br, int ldb) {
    v16bf f;
#pragma unroll
    for (int s = 0; s < 16; ++s)
        f[s] = (__bf16)br[(size_t)(s < 8 ? s : s + 8) * ldb];
    return f;
}

// B fragment from transposed operand (Bt stored N x K row-major):
// element e = Bt[(col0+e)*ldb + k0+lane]. Strided scalar loads, unconditional.
__device__ __forceinline__ v16bf frag_bt_row(const float* __restrict__ bt, int ldb) {
    v16bf f;
#pragma unroll
    for (int e = 0; e < 16; ++e)
        f[e] = (__bf16)bt[(size_t)e * ldb];
    return f;
}

__device__ __forceinline__ void epilogue(v8f acc, float* __restrict__ C, int ldc,
                                         int row0, int col0, const float* __restrict__ bias,
                                         int M, int N, int flags, int lane) {
    int n = col0 + (lane & 15);
    if (n >= N) return;
    int rbase = row0 + ((lane >> 4) << 3);
    float bv = (flags & F_BIAS) ? bias[n] : 0.0f;
#pragma unroll
    for (int r = 0; r < 8; ++r) {
        int m = rbase + r;
        if (m >= M) continue;
        float val = acc[r] + bv;
        if (flags & F_GELU) val = gelu_tanh(val);
        if (flags & F_ACCUM) C[(size_t)m * ldc + n] += val;
        else                 C[(size_t)m * ldc + n]  = val;
    }
}

// Generic batched WMMA GEMM: C = act(A @ B + bias).  256 thr = 8 waves,
// workgroup tile 128x64, each wave 32x32 (4 accumulators), K step 32.
// Requires: N % 64 == 0 (callers guarantee), K % 32 == 0. M arbitrary.
__global__ __launch_bounds__(256) void gemm_wmma_kernel(
        const float* __restrict__ A, int lda,
        const float* __restrict__ B, int ldb,
        const float* __restrict__ bias,
        float* __restrict__ C, int ldc,
        int M, int N, int K, int flags,
        long sA, long sB, long sC) {
    int z = blockIdx.z;
    A += (long)z * sA; B += (long)z * sB; C += (long)z * sC;
    int w    = threadIdx.x >> 5;
    int lane = threadIdx.x & 31;
    int row0 = blockIdx.y * 128 + (w >> 1) * 32;
    int col0 = blockIdx.x * 64  + (w & 1)  * 32;
    if (row0 >= M || col0 >= N) return;   // wave-uniform exit; EXEC all-1 inside

    // Clamp A row indices once; padded rows produce garbage only in output
    // rows >= M which the epilogue never stores.
    int mlo = row0 + (lane & 15);        if (mlo >= M) mlo = M - 1;
    int mhi = row0 + 16 + (lane & 15);   if (mhi >= M) mhi = M - 1;
    int hb  = (lane >> 4) << 3;          // 0 or 8
    const float* arow0 = A + (size_t)mlo * lda;
    const float* arow1 = A + (size_t)mhi * lda;

    const float* pb0;
    const float* pb1;
    long bstep;
    if (flags & F_TRANSB) {              // Bt is N x K row-major
        pb0 = B + (size_t)(col0 + (lane & 15)) * ldb;
        pb1 = B + (size_t)(col0 + 16 + (lane & 15)) * ldb;
        bstep = 32;                       // advance along K
    } else {                              // B is K x N row-major
        pb0 = B + (size_t)hb * ldb + col0 + (lane & 15);
        pb1 = pb0 + 16;
        bstep = (long)32 * ldb;           // advance 32 K-rows
    }

    v8f a00 = {0.f,0.f,0.f,0.f,0.f,0.f,0.f,0.f};
    v8f a01 = a00, a10 = a00, a11 = a00;

    for (int k0 = 0; k0 < K; k0 += 32) {
        v16bf fa0 = frag_a_row(arow0, k0, hb);
        v16bf fa1 = frag_a_row(arow1, k0, hb);
        v16bf fb0, fb1;
        if (flags & F_TRANSB) {
            fb0 = frag_a_row(pb0, 0, hb);
            fb1 = frag_a_row(pb1, 0, hb);
        } else {
            fb0 = frag_b_row(pb0, ldb);
            fb1 = frag_b_row(pb1, ldb);
        }
        pb0 += bstep; pb1 += bstep;
        a00 = __builtin_amdgcn_wmma_f32_16x16x32_bf16(false, fa0, false, fb0, (short)0, a00, false, false);
        a01 = __builtin_amdgcn_wmma_f32_16x16x32_bf16(false, fa0, false, fb1, (short)0, a01, false, false);
        a10 = __builtin_amdgcn_wmma_f32_16x16x32_bf16(false, fa1, false, fb0, (short)0, a10, false, false);
        a11 = __builtin_amdgcn_wmma_f32_16x16x32_bf16(false, fa1, false, fb1, (short)0, a11, false, false);
    }
    epilogue(a00, C, ldc, row0,      col0,      bias, M, N, flags, lane);
    epilogue(a01, C, ldc, row0,      col0 + 16, bias, M, N, flags, lane);
    epilogue(a10, C, ldc, row0 + 16, col0,      bias, M, N, flags, lane);
    epilogue(a11, C, ldc, row0 + 16, col0 + 16, bias, M, N, flags, lane);
}

// Fused 'observe' attention: one workgroup per (gripper g, head h).
// logits[s] = (q_gh . k_s + qWe_gh . ea_{s,g}) / 8 ; softmax over 2048 scene
// rows ; msg = (sum_w v + (sum_w ea) @ We_h) / (den + 1e-9).
__global__ __launch_bounds__(256) void edge_attn_observe_kernel(
        const float* __restrict__ q,   const float* __restrict__ qWe,
        const float* __restrict__ k,   const float* __restrict__ v,
        const float* __restrict__ ea,  const float* __restrict__ We,
        float* __restrict__ msgh) {
    __shared__ float s_logit[2048];
    __shared__ float s_q[64], s_qw[64];
    __shared__ float red[256];
    __shared__ float s_acc[2][4][64];
    int g  = blockIdx.x >> 3;
    int hh = blockIdx.x & 7;
    int t  = threadIdx.x;

    if (t < 64) {
        s_q[t]  = q  [(size_t)g * 512 + hh * 64 + t];
        s_qw[t] = qWe[(size_t)g * 512 + hh * 64 + t];
    }
    __syncthreads();

    for (int s = t; s < 2048; s += 256) {
        const float* kr = k  + (size_t)s * 512 + hh * 64;
        const float* er = ea + ((size_t)s * 64 + g) * 64;
        __builtin_prefetch(er + 16384, 0, 1);   // ea row of scene s+4, same gripper
        float acc = 0.0f;
#pragma unroll 8
        for (int d = 0; d < 64; ++d) acc += s_q[d] * kr[d] + s_qw[d] * er[d];
        s_logit[s] = acc * 0.125f;              // 1/sqrt(64)
    }
    __syncthreads();

    float lm = -3.4e38f;
    for (int s = t; s < 2048; s += 256) lm = fmaxf(lm, s_logit[s]);
    red[t] = lm; __syncthreads();
    for (int o = 128; o > 0; o >>= 1) { if (t < o) red[t] = fmaxf(red[t], red[t + o]); __syncthreads(); }
    float mx = red[0]; __syncthreads();

    float ls = 0.0f;
    for (int s = t; s < 2048; s += 256) { float w = __expf(s_logit[s] - mx); s_logit[s] = w; ls += w; }
    red[t] = ls; __syncthreads();
    for (int o = 128; o > 0; o >>= 1) { if (t < o) red[t] += red[t + o]; __syncthreads(); }
    float invden = 1.0f / (red[0] + 1e-9f);
    __syncthreads();

    int d = t & 63, grp = t >> 6;
    float av = 0.0f, aw = 0.0f;
    for (int s = grp; s < 2048; s += 4) {
        float w = s_logit[s];
        av += w * v[(size_t)s * 512 + hh * 64 + d];
        aw += w * ea[((size_t)s * 64 + g) * 64 + d];
    }
    s_acc[0][grp][d] = av;
    s_acc[1][grp][d] = aw;
    __syncthreads();
    if (t < 64) {
        s_q[t]  = s_acc[0][0][t] + s_acc[0][1][t] + s_acc[0][2][t] + s_acc[0][3][t];
        s_qw[t] = s_acc[1][0][t] + s_acc[1][1][t] + s_acc[1][2][t] + s_acc[1][3][t];
    }
    __syncthreads();
    if (t < 64) {
        float me = 0.0f;
#pragma unroll 8
        for (int e = 0; e < 64; ++e) me += s_qw[e] * We[(size_t)e * 512 + hh * 64 + t];
        msgh[(size_t)g * 512 + hh * 64 + t] = (s_q[t] + me) * invden;
    }
}

// Single-edge-per-dst types (guide/attend): msg row = ((h_lang@Wv + lpe@We)@Wo)*alpha
__global__ __launch_bounds__(512) void lang_msg_kernel(
        const float* __restrict__ h_lang, const float* __restrict__ lpe,
        const float* __restrict__ Wv, const float* __restrict__ We,
        const float* __restrict__ Wo, float* __restrict__ out_row) {
    __shared__ float tmp[512];
    __shared__ float hl[512];
    __shared__ float lp[64];
    int t = threadIdx.x;
    hl[t] = h_lang[t];
    if (t < 64) lp[t] = lpe[t];
    __syncthreads();
    float a = 0.0f;
    for (int i = 0; i < 512; ++i) a += hl[i] * Wv[(size_t)i * 512 + t];
    for (int e = 0; e < 64; ++e)  a += lp[e] * We[(size_t)e * 512 + t];
    tmp[t] = a;
    __syncthreads();
    float o = 0.0f;
    for (int i = 0; i < 512; ++i) o += tmp[i] * Wo[(size_t)i * 512 + t];
    out_row[t] = o * (1.0f / (1.0f + 1e-9f));   // alpha for single-edge softmax
}

// out = LayerNorm(h + msgA + msgB) ; one row (512 cols) per workgroup.
__global__ __launch_bounds__(256) void add_ln_kernel(
        const float* __restrict__ h,
        const float* __restrict__ msgA, int strideA,
        const float* __restrict__ msgB, int strideB,
        const float* __restrict__ gamma, const float* __restrict__ beta,
        float* __restrict__ out) {
    __shared__ float xs[512];
    __shared__ float red[256];
    int row = blockIdx.x, t = threadIdx.x;
    float local = 0.0f;
    for (int c = t; c < 512; c += 256) {
        float x = h[(size_t)row * 512 + c];
        if (msgA) x += msgA[(size_t)row * strideA + c];
        if (msgB) x += msgB[(size_t)row * strideB + c];
        xs[c] = x; local += x;
    }
    red[t] = local; __syncthreads();
    for (int o = 128; o > 0; o >>= 1) { if (t < o) red[t] += red[t + o]; __syncthreads(); }
    float mu = red[0] * (1.0f / 512.0f); __syncthreads();
    float lv = 0.0f;
    for (int c = t; c < 512; c += 256) { float d = xs[c] - mu; lv += d * d; }
    red[t] = lv; __syncthreads();
    for (int o = 128; o > 0; o >>= 1) { if (t < o) red[t] += red[t + o]; __syncthreads(); }
    float rstd = rsqrtf(red[0] * (1.0f / 512.0f) + 1e-5f);
    for (int c = t; c < 512; c += 256)
        out[(size_t)row * 512 + c] = (xs[c] - mu) * rstd * gamma[c] + beta[c];
}

__global__ void copy_f32_kernel(float* __restrict__ dst, const float* __restrict__ src, int n) {
    int i = blockIdx.x * 256 + threadIdx.x;
    if (i < n) dst[i] = src[i];
}

// ---------------------------------------------------------------------------
// Host side
// ---------------------------------------------------------------------------
namespace {
struct AttnP  { const float *Wq, *Wk, *Wv, *We, *Wo; };
struct NodeP  { const float *ln1_g, *ln1_b, *W1, *b1, *W2, *b2, *ln2_g, *ln2_b; };
struct LayerP { AttnP observe, guide, attend; NodeP scene, gripper, lang; };

inline void gemm(hipStream_t st, const float* A, int lda, const float* B, int ldb,
                 const float* bias, float* C, int ldc, int M, int N, int K,
                 int flags, int batch = 1, long sA = 0, long sB = 0, long sC = 0) {
    dim3 grid((unsigned)((N + 63) / 64), (unsigned)((M + 127) / 128), (unsigned)batch);
    gemm_wmma_kernel<<<grid, dim3(256, 1, 1), 0, st>>>(A, lda, B, ldb, bias, C, ldc,
                                                       M, N, K, flags, sA, sB, sC);
}
} // namespace

extern "C" void kernel_launch(void* const* d_in, const int* in_sizes, int n_in,
                              void* d_out, int out_size, void* d_ws, size_t ws_size,
                              hipStream_t stream) {
    (void)in_sizes; (void)out_size; (void)ws_size;
    const float* scene_feat   = (const float*)d_in[0];
    const float* gripper_feat = (const float*)d_in[2];
    const float* lang_feat    = (const float*)d_in[4];
    const float* edge_attr    = (const float*)d_in[6];
    // scene_pos (1), gripper_pos (3), edge_index (5) are dead in the reference.

    // Parameters: either one flat blob (n_in==8) or one pointer per leaf, both
    // in dict-insertion order of setup_inputs().
    int leaf = 7;
    const float* blob = (n_in > 8) ? nullptr : (const float*)d_in[7];
    auto next = [&](size_t n) -> const float* {
        if (n_in > 8) return (const float*)d_in[leaf++];
        const float* r = blob; blob += n; return r;
    };
    const float* scene_proj_w = next(512 * 512);
    const float* scene_proj_b = next(512);
    const float* grip_proj_w  = next(128 * 512);
    const float* grip_proj_b  = next(512);
    const float* lpe          = next(64);
    LayerP L[2];
    for (int l = 0; l < 2; ++l) {
        AttnP* at[3] = { &L[l].observe, &L[l].guide, &L[l].attend };
        for (int a = 0; a < 3; ++a) {
            at[a]->Wq = next(512 * 512); at[a]->Wk = next(512 * 512);
            at[a]->Wv = next(512 * 512); at[a]->We = next(64 * 512);
            at[a]->Wo = next(512 * 512);
        }
        NodeP* nd[3] = { &L[l].scene, &L[l].gripper, &L[l].lang };
        for (int nI = 0; nI < 3; ++nI) {
            nd[nI]->ln1_g = next(512);        nd[nI]->ln1_b = next(512);
            nd[nI]->W1    = next(512 * 2048); nd[nI]->b1    = next(2048);
            nd[nI]->W2    = next(2048 * 512); nd[nI]->b2    = next(512);
            nd[nI]->ln2_g = next(512);        nd[nI]->ln2_b = next(512);
        }
    }

    // Workspace carve (fp32), ~39 MB total.
    float* p = (float*)d_ws;
    auto carve = [&](size_t n) { float* r = p; p += n; return r; };
    float* h_scene    = carve(2048 * 512);
    float* h_grip     = carve(64 * 512);
    float* h_lang     = carve(512);
    float* qb         = carve(64 * 512);
    float* kb         = carve(2048 * 512);
    float* vb         = carve(2048 * 512);
    float* qWe        = carve(64 * 512);
    float* msgh       = carve(64 * 512);
    float* msg_grip   = carve(64 * 512);
    float* guide_row  = carve(512);
    float* attend_row = carve(512);
    float* x_s        = carve(2048 * 512);
    float* t1         = carve(2048 * 2048);
    float* t2         = carve(2048 * 512);
    float* x_g        = carve(64 * 512);
    float* t1g        = carve(64 * 2048);
    float* t2g        = carve(64 * 512);
    float* x_l        = carve(512);
    float* t1l        = carve(2048);
    float* t2l        = carve(512);

    // Input projections
    gemm(stream, scene_feat, 512, scene_proj_w, 512, scene_proj_b, h_scene, 512,
         2048, 512, 512, F_BIAS);
    gemm(stream, gripper_feat, 128, grip_proj_w, 512, grip_proj_b, h_grip, 512,
         64, 512, 128, F_BIAS);
    copy_f32_kernel<<<2, 256, 0, stream>>>(h_lang, lang_feat, 512);

    for (int l = 0; l < 2; ++l) {
        const LayerP& lp = L[l];
        // --- observe attention (scene -> gripper), dense 2048x64 bipartite ---
        gemm(stream, h_grip, 512, lp.observe.Wq, 512, nullptr, qb, 512, 64, 512, 512, 0);
        gemm(stream, h_scene, 512, lp.observe.Wk, 512, nullptr, kb, 512, 2048, 512, 512, 0);
        gemm(stream, h_scene, 512, lp.observe.Wv, 512, nullptr, vb, 512, 2048, 512, 512, 0);
        // qWe[g,h,:] = We_h^T q_gh : batched 64x64x64 GEMM over 8 heads (transB)
        gemm(stream, qb, 512, lp.observe.We, 512, nullptr, qWe, 512,
             64, 64, 64, F_TRANSB, 8, 64, 64, 64);
        edge_attn_observe_kernel<<<512, 256, 0, stream>>>(qb, qWe, kb, vb,
                                                          edge_attr, lp.observe.We, msgh);
        gemm(stream, msgh, 512, lp.observe.Wo, 512, nullptr, msg_grip, 512, 64, 512, 512, 0);

        // --- guide (lang -> gripper) and attend (lang -> scene): broadcast rows ---
        lang_msg_kernel<<<1, 512, 0, stream>>>(h_lang, lpe, lp.guide.Wv,
                                               lp.guide.We, lp.guide.Wo, guide_row);
        lang_msg_kernel<<<1, 512, 0, stream>>>(h_lang, lpe, lp.attend.Wv,
                                               lp.attend.We, lp.attend.Wo, attend_row);

        // --- node updates: x = LN(h+msg); ff = gelu(xW1+b1)W2+b2; h = LN(x+ff) ---
        // scene
        add_ln_kernel<<<2048, 256, 0, stream>>>(h_scene, attend_row, 0, nullptr, 0,
                                                lp.scene.ln1_g, lp.scene.ln1_b, x_s);
        gemm(stream, x_s, 512, lp.scene.W1, 2048, lp.scene.b1, t1, 2048,
             2048, 2048, 512, F_BIAS | F_GELU);
        gemm(stream, t1, 2048, lp.scene.W2, 512, lp.scene.b2, t2, 512,
             2048, 512, 2048, F_BIAS);
        add_ln_kernel<<<2048, 256, 0, stream>>>(x_s, t2, 512, nullptr, 0,
                                                lp.scene.ln2_g, lp.scene.ln2_b, h_scene);
        // gripper
        add_ln_kernel<<<64, 256, 0, stream>>>(h_grip, msg_grip, 512, guide_row, 0,
                                              lp.gripper.ln1_g, lp.gripper.ln1_b, x_g);
        gemm(stream, x_g, 512, lp.gripper.W1, 2048, lp.gripper.b1, t1g, 2048,
             64, 2048, 512, F_BIAS | F_GELU);
        gemm(stream, t1g, 2048, lp.gripper.W2, 512, lp.gripper.b2, t2g, 512,
             64, 512, 2048, F_BIAS);
        add_ln_kernel<<<64, 256, 0, stream>>>(x_g, t2g, 512, nullptr, 0,
                                              lp.gripper.ln2_g, lp.gripper.ln2_b, h_grip);
        // lang (msg == 0)
        add_ln_kernel<<<1, 256, 0, stream>>>(h_lang, nullptr, 0, nullptr, 0,
                                             lp.lang.ln1_g, lp.lang.ln1_b, x_l);
        gemm(stream, x_l, 512, lp.lang.W1, 2048, lp.lang.b1, t1l, 2048,
             1, 2048, 512, F_BIAS | F_GELU);
        gemm(stream, t1l, 2048, lp.lang.W2, 512, lp.lang.b2, t2l, 512,
             1, 512, 2048, F_BIAS);
        add_ln_kernel<<<1, 256, 0, stream>>>(x_l, t2l, 512, nullptr, 0,
                                             lp.lang.ln2_g, lp.lang.ln2_b, h_lang);
    }

    copy_f32_kernel<<<(64 * 512 + 255) / 256, 256, 0, stream>>>((float*)d_out, h_grip, 64 * 512);
}
